// CategoricalCrossEntropy_7756710936824
// MI455X (gfx1250) — matrix operand order repaired and checked by reading, not compiled, not run on hardware
//
#include <hip/hip_runtime.h>
#include <hip/hip_bf16.h>

#define NROWS 524288
#define DIN   64
#define HDIM  128
#define NB    256
#define ROWS_PER_BLK 128
#define NBLK  (NROWS / ROWS_PER_BLK)   // 4096 row-blocks
#define GRID  1024
#define THREADS 256

typedef __attribute__((ext_vector_type(16))) __bf16 v16bf;
typedef __attribute__((ext_vector_type(8)))  __bf16 v8bf;
typedef __attribute__((ext_vector_type(8)))  float  v8f;

__device__ __forceinline__ v8bf cvt8(const float4 a, const float4 b) {
    v8bf r;
    r[0] = (__bf16)a.x; r[1] = (__bf16)a.y; r[2] = (__bf16)a.z; r[3] = (__bf16)a.w;
    r[4] = (__bf16)b.x; r[5] = (__bf16)b.y; r[6] = (__bf16)b.z; r[7] = (__bf16)b.w;
    return r;
}
__device__ __forceinline__ v16bf cat16(v8bf lo, v8bf hi) {
    return __builtin_shufflevector(lo, hi, 0,1,2,3,4,5,6,7,8,9,10,11,12,13,14,15);
}

// branchless tanh-form GELU: 0.5x(1+tanh(0.79788456(x+0.044715x^3))) == x*sigmoid(2u)
__device__ __forceinline__ float gelu_fast(float x) {
    float u2 = x * x;
    float u  = x * fmaf(u2, 0.071354816f, 1.5957691216057308f);   // 2*0.79788456*(1+0.044715x^2)
    return x * __builtin_amdgcn_rcpf(1.0f + __expf(-u));
}

__global__ void zero_out_kernel(float* out) { *out = 0.0f; }

__global__ void __launch_bounds__(THREADS)
mlp_logscore_kernel(const float* __restrict__ X,  const float* __restrict__ Y,
                    const float* __restrict__ W1, const float* __restrict__ b1,
                    const float* __restrict__ W2, const float* __restrict__ b2,
                    float* __restrict__ out)
{
    // LDS: W2^T bf16 (64KB) + W1^T bf16 (16KB) + per-wave h staging (32KB)
    __shared__ __align__(16) __bf16 w2t[NB * HDIM];       // [n][k] 256x128
    __shared__ __align__(16) __bf16 w1t[HDIM * DIN];      // [n][k] 128x64
    __shared__ __align__(16) __bf16 hbuf[8 * 16 * HDIM];  // 8 waves x 16 rows x 128
    __shared__ float sAcc;

    const int tid = threadIdx.x;
    if (tid == 0) sAcc = 0.0f;

    // ---- stage weights once per block: transpose + f32->bf16 ----
    for (int idx = tid; idx < DIN * HDIM; idx += THREADS) {
        int k = idx >> 7, n = idx & (HDIM - 1);            // W1 is [k][n] row-major
        w1t[n * DIN + k] = (__bf16)W1[idx];
    }
    for (int idx = tid; idx < HDIM * NB; idx += THREADS) {
        int k = idx >> 8, n = idx & (NB - 1);              // W2 is [k][n] row-major
        w2t[n * HDIM + k] = (__bf16)W2[idx];
    }
    __syncthreads();   // weights visible to all waves; only barrier before the final reduce

    const int wave = tid >> 5;
    const int lane = tid & 31;
    const int l16  = lane & 15;
    const int hi   = lane >> 4;       // lane half: K sub-range (A) / K half (B)
    const int kb   = hi * 8;          // A-fragment K base per ISA layout
    __bf16* hw = hbuf + wave * (16 * HDIM);   // per-wave private staging: no barriers needed

    // biases depend only on (tile, l16): hoist out of the row loop
    float bias1[8], bias2[16];
    #pragma unroll
    for (int t = 0; t < 8; t++)  bias1[t] = b1[t * 16 + l16];
    #pragma unroll
    for (int t = 0; t < 16; t++) bias2[t] = b2[t * 16 + l16];

    float waveAcc = 0.0f;   // per-wave running score sum across all row-blocks

    for (int rb = blockIdx.x; rb < NBLK; rb += gridDim.x) {
        const int base = rb * ROWS_PER_BLK;

        // prefetch next row-block of X (global_prefetch_b8)
        {
            int nrb = rb + gridDim.x;
            if (nrb < NBLK) {
                const float* pf = X + ((size_t)(nrb * ROWS_PER_BLK + (tid >> 1))) * DIN
                                    + (size_t)(tid & 1) * 32;
                __builtin_prefetch(pf, 0, 1);
            }
        }

        // ================= GEMM1: h = gelu(X*W1 + b1), 16 rows per wave =================
        const int row = base + wave * 16 + l16;            // A layout: lane = M
        const float* xr = X + (size_t)row * DIN + kb;
        v16bf a0 = cat16(cvt8(*(const float4*)(xr +  0), *(const float4*)(xr +  4)),
                         cvt8(*(const float4*)(xr + 16), *(const float4*)(xr + 20)));
        v16bf a1 = cat16(cvt8(*(const float4*)(xr + 32), *(const float4*)(xr + 36)),
                         cvt8(*(const float4*)(xr + 48), *(const float4*)(xr + 52)));

        #pragma unroll
        for (int t = 0; t < 8; t++) {                      // 8 tiles of 16 cols -> H=128
            const int n = t * 16 + l16;                    // B layout: lane = N
            const __bf16* bp = w1t + n * DIN + hi * 8;     // K = 8*hi+{0..7}, 16+8*hi+{0..7} (ISA table order)
            v16bf bf0 = cat16(*(const v8bf*)(bp +  0), *(const v8bf*)(bp + 16));
            v16bf bf1 = cat16(*(const v8bf*)(bp + 32), *(const v8bf*)(bp + 48));
            v8f c = {};
            c = __builtin_amdgcn_wmma_f32_16x16x32_bf16(false, a0, false, bf0, (short)0, c, false, false);
            c = __builtin_amdgcn_wmma_f32_16x16x32_bf16(false, a1, false, bf1, (short)0, c, false, false);
            #pragma unroll
            for (int i = 0; i < 8; i++) {                  // C layout: elem i -> M = i + 8*hi
                float x = gelu_fast(c[i] + bias1[t]);
                hw[(i + 8 * hi) * HDIM + t * 16 + l16] = (__bf16)x;
            }
        }
        // per-wave RAW on hbuf handled by in-order LDS + s_wait_dscnt (no barrier)

        // ================= GEMM2: logits = h*W2 + b2, streamed softmax =================
        v16bf ha[4];
        #pragma unroll
        for (int kc = 0; kc < 4; kc++) {                   // K = 128 in 4 chunks of 32
            const __bf16* p = hw + l16 * HDIM + kc * 32 + kb;
            ha[kc] = cat16(*(const v8bf*)p, *(const v8bf*)(p + 16));
        }

        int ybin[8];
        #pragma unroll
        for (int i = 0; i < 8; i++) {                      // bins for the 8 rows of this half
            float y = Y[base + wave * 16 + i + 8 * hi];
            int bb = (int)ceilf(y * 256.0f) - 1;           // searchsorted(left) - 1
            ybin[i] = bb < 0 ? 0 : (bb > 255 ? 255 : bb);
        }

        float vmax[8], vsum[8], vsel[8];
        #pragma unroll
        for (int i = 0; i < 8; i++) { vmax[i] = -3.4e38f; vsum[i] = 0.0f; vsel[i] = 0.0f; }

        #pragma unroll 2
        for (int t = 0; t < 16; t++) {                     // 16 tiles of 16 cols -> B=256
            v8f c = {};
            const __bf16* bp = w2t + (t * 16 + l16) * HDIM + hi * 8;
            #pragma unroll
            for (int kc = 0; kc < 4; kc++) {
                v16bf bfg = cat16(*(const v8bf*)(bp + kc * 32), *(const v8bf*)(bp + kc * 32 + 16));
                c = __builtin_amdgcn_wmma_f32_16x16x32_bf16(false, ha[kc], false, bfg, (short)0, c, false, false);
            }
            const int n = t * 16 + l16;
            #pragma unroll
            for (int i = 0; i < 8; i++) {                  // online per-lane softmax
                float x  = c[i] + bias2[t];
                float nm = fmaxf(vmax[i], x);
                vsum[i]  = vsum[i] * __expf(vmax[i] - nm) + __expf(x - nm);
                vmax[i]  = nm;
                if (n == ybin[i]) vsel[i] = x;             // capture logit at target bin
            }
        }

        // merge 16-lane groups (max/sum rescale + sel gather) and score
        float part = 0.0f;
        #pragma unroll
        for (int i = 0; i < 8; i++) {
            float mx = vmax[i], sm = vsum[i], sl = vsel[i];
            #pragma unroll
            for (int m = 1; m < 16; m <<= 1) {
                float om = __shfl_xor(mx, m);
                float os = __shfl_xor(sm, m);
                float ol = __shfl_xor(sl, m);
                float nm = fmaxf(mx, om);
                sm = sm * __expf(mx - nm) + os * __expf(om - nm);
                mx = nm;
                sl += ol;
            }
            float mass = __expf(sl - mx) * __builtin_amdgcn_rcpf(sm);  // softmax prob at bin
            part += -__logf(fmaf(mass, 256.0f, 1e-45f));   // density = mass / (1/256)
        }
        if (l16 != 0) part = 0.0f;                         // one contributor per 16-lane group
        #pragma unroll
        for (int m = 1; m < 32; m <<= 1) part += __shfl_xor(part, m);
        waveAcc += part;                                   // same value in all lanes
    }

    if (lane == 0) atomicAdd(&sAcc, waveAcc);              // one ds_add_f32 per wave
    __syncthreads();
    if (tid == 0) atomicAdd(out, sAcc);                    // one global_atomic_add_f32 per block
}

extern "C" void kernel_launch(void* const* d_in, const int* in_sizes, int n_in,
                              void* d_out, int out_size, void* d_ws, size_t ws_size,
                              hipStream_t stream) {
    const float* X  = (const float*)d_in[0];
    const float* Y  = (const float*)d_in[1];
    const float* W1 = (const float*)d_in[2];
    const float* b1 = (const float*)d_in[3];
    const float* W2 = (const float*)d_in[4];
    const float* b2 = (const float*)d_in[5];
    float* out = (float*)d_out;

    zero_out_kernel<<<1, 1, 0, stream>>>(out);
    mlp_logscore_kernel<<<GRID, THREADS, 0, stream>>>(X, Y, W1, b1, W2, b2, out);
}
